// DifferentiableDAG_21818433864163
// MI455X (gfx1250) — matrix operand order repaired and checked by reading, not compiled, not run on hardware
//
#include <hip/hip_runtime.h>

#define Bn 4096
#define Vn 512
#define Hn 64

typedef __bf16 v16bf __attribute__((ext_vector_type(16)));
typedef float  v8f   __attribute__((ext_vector_type(8)));

struct Q2 { uint4 a, b; };   // 32B: two 16B-aligned LDS b128 loads -> one fragment

// fp32 -> bf16 bits, round-to-nearest-even (finite inputs)
__device__ __forceinline__ unsigned int bf16_rne(float x) {
    unsigned int u = __builtin_bit_cast(unsigned int, x);
    u += 0x7FFFu + ((u >> 16) & 1u);
    return u >> 16;
}

// split x ~= hi + lo, both bf16 (bf16x3 emulation of fp32 GEMM)
__device__ __forceinline__ void bsplit(float x, unsigned int& h, unsigned int& l) {
    h = bf16_rne(x);
    float hf = __builtin_bit_cast(float, h << 16);
    l = bf16_rne(x - hf);
}

// pack two bf16-hi/lo pairs into dwords (element k in low half, k+1 in high half)
__device__ __forceinline__ void pack2(float x0, float x1, unsigned int& ph, unsigned int& pl) {
    unsigned int h0, l0, h1, l1;
    bsplit(x0, h0, l0);
    bsplit(x1, h1, l1);
    ph = h0 | (h1 << 16);
    pl = l0 | (l1 << 16);
}

__device__ __forceinline__ void async_b128(unsigned ldsByteOff, const unsigned* sbase,
                                           unsigned gByteOff) {
    asm volatile("global_load_async_to_lds_b128 %0, %1, %2 offset:0"
                 :
                 : "v"(ldsByteOff), "v"(gByteOff), "s"(sbase)
                 : "memory");
}

// ---------------------------------------------------------------------------
// Kernel 1: hard adjacency. adj[v,c] = (sigmoid(logit)>0.5) && v!=c
// ---------------------------------------------------------------------------
__global__ __launch_bounds__(256) void dag_adj_kernel(
    const float* __restrict__ logits, float* __restrict__ adj)
{
    int idx = blockIdx.x * 256 + threadIdx.x;   // V*V
    int v = idx >> 9;
    int c = idx & (Vn - 1);
    adj[idx] = (logits[idx] > 0.0f && v != c) ? 1.0f : 0.0f;
}

// ---------------------------------------------------------------------------
// Prep X: pack X [B,V] into bf16 hi/lo K-pair planes, layout [b][kp] (kp=v/2).
// ---------------------------------------------------------------------------
__global__ __launch_bounds__(256) void dag_packx_kernel(
    const float* __restrict__ X, unsigned* __restrict__ xh, unsigned* __restrict__ xl)
{
    int p = blockIdx.x * 256 + threadIdx.x;     // B * V/2 pairs
    int b  = p >> 8;
    int kp = p & 255;
    const float* s = X + (size_t)b * Vn + 2 * kp;
    pack2(s[0], s[1], xh[p], xl[p]);
}

// ---------------------------------------------------------------------------
// Prep W: pack adjacency-masked W1 into bf16 hi/lo planes, fragment-major:
//   wg[((i*16 + kchunk)*64 + n)*16 + kpl],  v = kchunk*32 + 2*kpl (+1)
// so the main kernel's per-(i,kchunk) W tile is one contiguous 4KB block.
// ---------------------------------------------------------------------------
__global__ __launch_bounds__(256) void dag_packw_kernel(
    const float* __restrict__ W1, const float* __restrict__ adj,
    unsigned* __restrict__ wh, unsigned* __restrict__ wl)
{
    int g = blockIdx.x * 256 + threadIdx.x;     // V * 16 * 64 * 16 dwords
    int i   = g >> 14;
    int r   = g & 16383;
    int kc  = r >> 10;
    int rem = r & 1023;
    int n   = rem >> 4;
    int kpl = rem & 15;
    int v0  = kc * 32 + 2 * kpl;
    float m0 = adj[(size_t)v0 * Vn + i];
    float m1 = adj[(size_t)(v0 + 1) * Vn + i];
    const float* s = W1 + ((size_t)i * Vn + v0) * Hn + n;
    pack2(s[0] * m0, s[Hn] * m1, wh[g], wl[g]);
}

// ---------------------------------------------------------------------------
// Main kernel: bf16x3 WMMA GEMM + relu + projection. Fill is a pure async
// global->LDS copy of pre-packed planes. Each wave computes 32(batch) x 64(H)
// for one net i: two A row-groups share every B fragment (24 WMMA : 24 DS).
// grid = (i=512, btile=16), block = 256 (8 waves, 256 batch rows).
// ---------------------------------------------------------------------------
__global__ __launch_bounds__(256) void dag_wmma_kernel(
    const unsigned* __restrict__ xh, const unsigned* __restrict__ xl,
    const unsigned* __restrict__ wh, const unsigned* __restrict__ wl,
    const float* __restrict__ b1,  const float* __restrict__ W2,
    const float* __restrict__ b2,  float* __restrict__ out)
{
    // X tile: [row][kpair], row stride 20 dwords (80B, 16B-aligned, bank-spread)
    __shared__ unsigned int xph[256][20], xpl[256][20];
    // W tile: [h][kpair] fragment-major, same stride
    __shared__ unsigned int wph[64][20],  wpl[64][20];

    const int i    = blockIdx.x;
    const int b0   = blockIdx.y * 256;
    const int tid  = threadIdx.x;
    const int wave = tid >> 5;
    const int lane = tid & 31;
    const int lrow = lane & 15;
    const int half = lane >> 4;

    // per-thread fill addressing (one X row per thread; 4 W dwords per thread)
    const unsigned xldsOff  = (unsigned)(uintptr_t)&xph[tid][0];
    const unsigned xldsOffL = (unsigned)(uintptr_t)&xpl[tid][0];
    const unsigned wn  = tid >> 2;
    const unsigned wqb = (tid & 3) * 4;
    const unsigned wldsOff  = (unsigned)(uintptr_t)&wph[wn][wqb];
    const unsigned wldsOffL = (unsigned)(uintptr_t)&wpl[wn][wqb];

    v8f acc[2][4] = {};   // 2 row-groups x 4 N-tiles of 16x16 f32

    for (int k0 = 0; k0 < Vn; k0 += 32) {
        const int kc = k0 >> 5;
        __syncthreads();

        // ---- async copy X tile [256 rows x 16 kpairs] ----
        {
            unsigned voff = ((((unsigned)(b0 + tid)) << 8) + (unsigned)(k0 >> 1)) << 2;
            async_b128(xldsOff,       xh, voff);
            async_b128(xldsOff + 16,  xh, voff + 16);
            async_b128(xldsOff + 32,  xh, voff + 32);
            async_b128(xldsOff + 48,  xh, voff + 48);
            async_b128(xldsOffL,      xl, voff);
            async_b128(xldsOffL + 16, xl, voff + 16);
            async_b128(xldsOffL + 32, xl, voff + 32);
            async_b128(xldsOffL + 48, xl, voff + 48);
        }
        // ---- async copy W tile (contiguous 4KB block per plane) ----
        {
            unsigned voff = ((((unsigned)i * 16 + kc) << 10) + ((unsigned)tid << 2)) << 2;
            async_b128(wldsOff,  wh, voff);
            async_b128(wldsOffL, wl, voff);
        }
        asm volatile("s_wait_asynccnt 0" ::: "memory");
        __syncthreads();

        // ---- A fragments: two row-groups, two b128 loads per plane ----
        v16bf ah[2], al[2];
        #pragma unroll
        for (int g = 0; g < 2; ++g) {
            const int rowA = wave * 32 + g * 16 + lrow;
            Q2 qah, qal;
            qah.a = *(const uint4*)&xph[rowA][half * 4];
            qah.b = *(const uint4*)&xph[rowA][8 + half * 4];
            qal.a = *(const uint4*)&xpl[rowA][half * 4];
            qal.b = *(const uint4*)&xpl[rowA][8 + half * 4];
            ah[g] = __builtin_bit_cast(v16bf, qah);
            al[g] = __builtin_bit_cast(v16bf, qal);
        }

        // ---- B fragments per N-tile, shared by both row-groups ----
        #pragma unroll
        for (int t = 0; t < 4; ++t) {
            const int nrow = t * 16 + lrow;
            Q2 qbh, qbl;
            qbh.a = *(const uint4*)&wph[nrow][half * 4];
            qbh.b = *(const uint4*)&wph[nrow][8 + half * 4];
            qbl.a = *(const uint4*)&wpl[nrow][half * 4];
            qbl.b = *(const uint4*)&wpl[nrow][8 + half * 4];
            v16bf bh = __builtin_bit_cast(v16bf, qbh);
            v16bf bl = __builtin_bit_cast(v16bf, qbl);
            #pragma unroll
            for (int g = 0; g < 2; ++g) {
                acc[g][t] = __builtin_amdgcn_wmma_f32_16x16x32_bf16(
                                false, ah[g], false, bh, (short)0, acc[g][t], false, false);
                acc[g][t] = __builtin_amdgcn_wmma_f32_16x16x32_bf16(
                                false, al[g], false, bh, (short)0, acc[g][t], false, false);
                acc[g][t] = __builtin_amdgcn_wmma_f32_16x16x32_bf16(
                                false, ah[g], false, bl, (short)0, acc[g][t], false, false);
            }
        }
    }

    // ---- epilogue: relu(+b1) * W2, reduce over H (16-lane groups), +b2 ----
    float b1v[4], w2v[4];
    #pragma unroll
    for (int t = 0; t < 4; ++t) {
        int h = t * 16 + lrow;
        b1v[t] = b1[i * Hn + h];
        w2v[t] = W2[i * Hn + h];
    }
    float bias2 = b2[i];

    #pragma unroll
    for (int g = 0; g < 2; ++g) {
        #pragma unroll
        for (int r = 0; r < 8; ++r) {
            float p = 0.0f;
            #pragma unroll
            for (int t = 0; t < 4; ++t) {
                float v = acc[g][t][r] + b1v[t];
                v = fmaxf(v, 0.0f);
                p += v * w2v[t];
            }
            p += __shfl_xor(p, 1, 32);
            p += __shfl_xor(p, 2, 32);
            p += __shfl_xor(p, 4, 32);
            p += __shfl_xor(p, 8, 32);
            if (lrow == 0) {
                int brow = b0 + wave * 32 + g * 16 + r + half * 8;  // M = r + 8*half
                out[(size_t)brow * Vn + i] = p + bias2;
            }
        }
    }
}

extern "C" void kernel_launch(void* const* d_in, const int* in_sizes, int n_in,
                              void* d_out, int out_size, void* d_ws, size_t ws_size,
                              hipStream_t stream) {
    const float* X      = (const float*)d_in[0];
    const float* logits = (const float*)d_in[1];
    const float* W1     = (const float*)d_in[2];
    const float* b1     = (const float*)d_in[3];
    const float* W2     = (const float*)d_in[4];
    const float* b2     = (const float*)d_in[5];

    float* out = (float*)d_out;
    float* adj = out + (size_t)Bn * Vn;   // second tuple output region

    // workspace layout (dwords): xh | xl | wh | wl  (~72 MB total)
    const size_t NXP = (size_t)Bn * (Vn / 2);          // 1,048,576
    const size_t NWP = (size_t)Vn * (Vn / 2) * Hn;     // 8,388,608
    unsigned* xh = (unsigned*)d_ws;
    unsigned* xl = xh + NXP;
    unsigned* wh = xl + NXP;
    unsigned* wl = wh + NWP;

    dag_adj_kernel<<<(Vn * Vn) / 256, 256, 0, stream>>>(logits, adj);
    dag_packx_kernel<<<NXP / 256, 256, 0, stream>>>(X, xh, xl);
    dag_packw_kernel<<<NWP / 256, 256, 0, stream>>>(W1, adj, wh, wl);

    dim3 grid(Vn, Bn / 256);
    dag_wmma_kernel<<<grid, 256, 0, stream>>>(xh, xl, wh, wl, b1, W2, b2, out);
}
